// ReDoTrAS_60464549593307
// MI455X (gfx1250) — compile-verified, matrix operand order checked by reading, not run on hardware
//
#include <hip/hip_runtime.h>

// Problem constants (from reference)
#define DIN   1024
#define HH    16
#define SEGL  512
#define SLEN  128
#define BB    2
#define SS    2048
#define LL    768   // SLEN + SEG + SLEN
#define NSEG  4

typedef _Float16 h8  __attribute__((ext_vector_type(8)));
typedef _Float16 h16 __attribute__((ext_vector_type(16)));
typedef float    f8  __attribute__((ext_vector_type(8)));

// ---------------------------------------------------------------------------
// WMMA helpers.
// Operand fragment for V_WMMA_F32_16X16X32_F16.
// A (16x32, row-major source): lane m = lane&15; per-lane halves hold
//   K = (h&7) + (h>=8 ? 16 : 0) + (lane>=16 ? 8 : 0)   [ISA 7.12.2]
// B (32x16) read from a matrix stored as M[n][k] (i.e. B^T row-major):
//   identical per-lane pattern with row = n = lane&15.
// ---------------------------------------------------------------------------
__device__ __forceinline__ h16 load_frag(const _Float16* rowPtr, int k0, int lane) {
  const int off = k0 + ((lane >> 4) << 3);
  h8 lo = *reinterpret_cast<const h8*>(rowPtr + off);
  h8 hi = *reinterpret_cast<const h8*>(rowPtr + off + 16);
  h16 r;
#pragma unroll
  for (int i = 0; i < 8; ++i) { r[i] = lo[i]; r[i + 8] = hi[i]; }
  return r;
}

__device__ __forceinline__ f8 wmma32(h16 a, h16 b, f8 c) {
  return __builtin_amdgcn_wmma_f32_16x16x32_f16(
      /*neg_a=*/false, a, /*neg_b=*/false, b,
      /*c_mod=*/(short)0, c, /*reuse_a=*/false, /*reuse_b=*/false);
}

// Async global -> LDS copy of one 16B chunk per lane (gfx1250 ASYNCcnt path).
// LDS address = low 32 bits of the generic pointer (flat->LDS aperture
// truncation, ISA 10.2).
__device__ __forceinline__ void async_b128_to_lds(const _Float16* g, _Float16* l) {
  asm volatile("global_load_async_to_lds_b128 %0, %1, off"
               :: "v"((unsigned)(unsigned long)l), "v"(g)
               : "memory");
}

// ---------------------------------------------------------------------------
// f32 -> f16 convert, with modulo on source for state broadcast
// ---------------------------------------------------------------------------
__global__ void k_cvt(const float* __restrict__ src, _Float16* __restrict__ dst,
                      long n, long srcMod) {
  long i = (long)blockIdx.x * blockDim.x + threadIdx.x;
  if (i < n) dst[i] = (_Float16)src[i % srcMod];
}

// W[h][k][n] (f32) -> Wt[h][n][k] (f16): transposed so the B operand reads rows.
__global__ void k_wT(const float* __restrict__ W, _Float16* __restrict__ Wt,
                     int Hn, int K, int N) {
  long i = (long)blockIdx.x * blockDim.x + threadIdx.x;
  long total = (long)Hn * K * N;
  if (i >= total) return;
  int n = (int)(i % N);
  long t = i / N;
  int k = (int)(t % K);
  int h = (int)(t / K);
  Wt[(long)h * K * N + (long)n * K + k] = (_Float16)W[i];
}

// ---------------------------------------------------------------------------
// Tile store: storeMode 0 = f16 row-major, 1 = f16 transposed (build V^T),
//             2 = f32 row-major (final outputs)
// ---------------------------------------------------------------------------
__device__ __forceinline__ void store_tile(f8 acc, void* Cv, long cB, long cH,
                                           int b, int h, int ldc, int row0,
                                           int n, int mode) {
  if (mode == 2) {
    float* C = (float*)Cv + (long)b * cB + (long)h * cH;
#pragma unroll
    for (int r = 0; r < 8; ++r) C[(long)(row0 + r) * ldc + n] = acc[r];
  } else if (mode == 1) {
    _Float16* C = (_Float16*)Cv + (long)b * cB + (long)h * cH;
#pragma unroll
    for (int r = 0; r < 8; ++r) C[(long)n * ldc + row0 + r] = (_Float16)acc[r];
  } else {
    _Float16* C = (_Float16*)Cv + (long)b * cB + (long)h * cH;
#pragma unroll
    for (int r = 0; r < 8; ++r) C[(long)(row0 + r) * ldc + n] = (_Float16)acc[r];
  }
}

// ---------------------------------------------------------------------------
// Per-head GEMM: C[b,h][M,N] = A[b,h][M,K] @ W[h][K,N]   (W given transposed)
// block = 256 (8 waves, each a 16-row strip).
// K >= 256 path: grid.x = (M/128)*(N/64); each block owns a 128x64 C tile.
//   The 64-col x 256-K weight strip is double-buffered in LDS via async
//   global->LDS copies (ASYNCcnt), overlapping the next slice's DMA with the
//   current slice's WMMAs. 4 accumulators/wave; A fragment reused 4x.
// K < 256 path: grid.x = M/128; direct global-load pipeline.
// ---------------------------------------------------------------------------
__global__ __launch_bounds__(256)
void k_gemm(const _Float16* __restrict__ A, long aB, long aH, int lda,
            const _Float16* __restrict__ Wt, long wH,
            int N, int K, int mBlocks,
            void* __restrict__ Cv, long cB, long cH, int ldc,
            int cRowOff, int storeMode) {
  __shared__ _Float16 wbuf[2][64 * 256];   // 2 x 32 KB weight strips
  const int tid  = threadIdx.x;
  const int lane = tid & 31;
  const int wave = tid >> 5;
  const int mb = blockIdx.x % mBlocks;
  const int nb = blockIdx.x / mBlocks;
  const int rowBase = mb * 128 + wave * 16;
  const int b = blockIdx.z, h = blockIdx.y;
  const int n16 = lane & 15;
  const int mofs = (lane >> 4) << 3;
  const int row0 = cRowOff + rowBase + mofs;

  const _Float16* Ap = A + (long)b * aB + (long)h * aH + (long)(rowBase + n16) * lda;
  const _Float16* Wp = Wt + (long)h * wH;

  if (K >= 256) {
    const int ng = nb * 64;                // this block's 64-col N group
    const int nk = K >> 8;                 // number of 256-wide K slices
    auto stage = [&](int i, int bufIdx) {  // async-stage slice i
      const _Float16* Wkk = Wp + i * 256;
      for (int c = tid; c < 64 * 32; c += 256) {  // 16B chunks, 8 per thread
        const int row = c >> 5;
        const int col = (c & 31) << 3;            // halves
        async_b128_to_lds(Wkk + (long)(ng + row) * K + col,
                          &wbuf[bufIdx][row * 256 + col]);
      }
    };
    stage(0, 0);
    f8 acc[4] = {};
    for (int i = 0; i < nk; ++i) {
      if (i + 1 < nk) {
        stage(i + 1, (i + 1) & 1);
        // 8 chunks/thread just issued may stay outstanding; ASYNCcnt completes
        // in order, so <=8 outstanding ==> slice i has fully landed.
        asm volatile("s_wait_asynccnt 0x8" ::: "memory");
      } else {
        asm volatile("s_wait_asynccnt 0x0" ::: "memory");
      }
      __syncthreads();
      const _Float16* wb = &wbuf[i & 1][0];
      const int kk = i << 8;
      for (int k0 = 0; k0 < 256; k0 += 32) {
        h16 af = load_frag(Ap, kk + k0, lane);
#pragma unroll
        for (int s = 0; s < 4; ++s) {
          h16 bf = load_frag(wb + (s * 16 + n16) * 256, k0, lane);
          acc[s] = wmma32(af, bf, acc[s]);
        }
      }
      __syncthreads();   // all waves done reading before this buffer is reused
    }
#pragma unroll
    for (int s = 0; s < 4; ++s)
      store_tile(acc[s], Cv, cB, cH, b, h, ldc, row0, ng + s * 16 + n16, storeMode);
  } else {
    for (int nt = 0; nt < N; nt += 16) {
      f8 acc = {};
      const _Float16* Wrow = Wp + (long)(nt + n16) * K;
      for (int k0 = 0; k0 < K; k0 += 32) {
        h16 af = load_frag(Ap, k0, lane);
        h16 bf = load_frag(Wrow, k0, lane);
        acc = wmma32(af, bf, acc);
      }
      store_tile(acc, Cv, cB, cH, b, h, ldc, row0, nt + n16, storeMode);
    }
  }
}

// ---------------------------------------------------------------------------
// Causal flash attention over L=768, head dim D (32 or 64).
// Q,K row-major (L,D); V given transposed (D,L). Output f16 row-major with
// arbitrary (base,ldo,head-col-offset) so attn3 can scatter into (B,L,H*64).
// grid = (L/64, H, B); block = 128; each wave owns 16 query rows.
// ---------------------------------------------------------------------------
template <int D>
__global__ __launch_bounds__(128)
void k_attn(const _Float16* __restrict__ Q, const _Float16* __restrict__ Kk,
            const _Float16* __restrict__ Vt, long bhStride,
            _Float16* __restrict__ Out, long outB, long outH, int ldo) {
  constexpr int NC = D / 16;            // output feature chunks
  __shared__ _Float16 pbuf[4][16 * 32]; // per-wave P staging (16 rows x 32 keys)

  const int lane = threadIdx.x & 31;
  const int wave = threadIdx.x >> 5;
  const int qbase = blockIdx.x * 64 + wave * 16;
  const int b = blockIdx.z, h = blockIdx.y;
  const int n16 = lane & 15;
  const int mofs = (lane >> 4) << 3;
  const float scale = 0.125f;           // 1/sqrt(DK=64), fixed in reference

  const long base = (long)(b * HH + h) * bhStride;
  const _Float16* Qp = Q + base;
  const _Float16* Kp = Kk + base;
  const _Float16* Vp = Vt + base;
  const _Float16* Qrow = Qp + (long)(qbase + n16) * D;
  _Float16* plds = &pbuf[wave][0];

  f8 o[NC] = {};
  float m_i[8], l_i[8];
#pragma unroll
  for (int r = 0; r < 8; ++r) { m_i[r] = -1e30f; l_i[r] = 0.0f; }

  const int kmax = qbase + 15;          // causal limit (inclusive)
  for (int kt = 0; kt <= kmax; kt += 32) {
    // ---- scores for keys [kt, kt+32) : two 16x16 tiles ----
    f8 s0 = {}, s1 = {};
#pragma unroll
    for (int k0 = 0; k0 < D; k0 += 32) {
      h16 qa  = load_frag(Qrow, k0, lane);
      h16 kb0 = load_frag(Kp + (long)(kt + n16) * D, k0, lane);
      h16 kb1 = load_frag(Kp + (long)(kt + 16 + n16) * D, k0, lane);
      s0 = wmma32(qa, kb0, s0);
      s1 = wmma32(qa, kb1, s1);
    }
    // ---- scale, causal mask, online softmax update ----
#pragma unroll
    for (int r = 0; r < 8; ++r) {
      const int qAbs = qbase + r + mofs;
      float v0 = s0[r] * scale; if (kt + n16 > qAbs)      v0 = -1e30f;
      float v1 = s1[r] * scale; if (kt + 16 + n16 > qAbs) v1 = -1e30f;
      float tmax = fmaxf(v0, v1);
#pragma unroll
      for (int sh = 1; sh < 16; sh <<= 1) tmax = fmaxf(tmax, __shfl_xor(tmax, sh, 16));
      const float mnew = fmaxf(m_i[r], tmax);
      const float corr = __expf(m_i[r] - mnew);
      m_i[r] = mnew;
      const float p0 = __expf(v0 - mnew);
      const float p1 = __expf(v1 - mnew);
      float rs = p0 + p1;
#pragma unroll
      for (int sh = 1; sh < 16; sh <<= 1) rs += __shfl_xor(rs, sh, 16);
      l_i[r] = l_i[r] * corr + rs;
#pragma unroll
      for (int c = 0; c < NC; ++c) o[c][r] *= corr;
      const int ml = r + mofs;
      plds[ml * 32 + n16]      = (_Float16)p0;
      plds[ml * 32 + 16 + n16] = (_Float16)p1;
    }
    asm volatile("" ::: "memory");  // keep P stores before fragment reloads
    // ---- P @ V : A = P(16x32) from LDS, B from V^T rows ----
    h16 pa = load_frag(plds + n16 * 32, 0, lane);
#pragma unroll
    for (int c = 0; c < NC; ++c) {
      h16 vb = load_frag(Vp + (long)(c * 16 + n16) * LL, kt, lane);
      o[c] = wmma32(pa, vb, o[c]);
    }
  }
  // ---- normalize and store ----
  _Float16* Op = Out + (long)b * outB + (long)h * outH;
#pragma unroll
  for (int c = 0; c < NC; ++c) {
#pragma unroll
    for (int r = 0; r < 8; ++r) {
      float val = o[c][r] / l_i[r];
      Op[(long)(qbase + r + mofs) * ldo + c * 16 + n16] = (_Float16)val;
    }
  }
}

// ---------------------------------------------------------------------------
// Host orchestration
// ---------------------------------------------------------------------------
extern "C" void kernel_launch(void* const* d_in, const int* in_sizes, int n_in,
                              void* d_out, int out_size, void* d_ws, size_t ws_size,
                              hipStream_t stream) {
  (void)in_sizes; (void)n_in; (void)out_size; (void)ws_size;

  const float* x   = (const float*)d_in[0];
  const float* st0 = (const float*)d_in[1];
  const float* W[24];
  for (int i = 0; i < 24; ++i) W[i] = (const float*)d_in[2 + i];
  float* out = (float*)d_out;

  enum { iWk = 0, iWq, iWv, iWk_st, iWq_st, iWv_st,
         iWdk, iWdk_ss, iWdq_ss, iWdv_ss, iWdk_se, iWdq_se, iWdv_se,
         iWuk, iWuq, iWuv, iWuk_ss, iWuq_ss, iWuv_ss, iWuk_se, iWuq_se, iWuv_se,
         iWout, iWout_st };

  // bump allocator over workspace
  char* ws = (char*)d_ws;
  size_t pos = 0;
  auto alloc = [&](size_t bytes) -> void* {
    pos = (pos + 255) & ~(size_t)255;
    void* p = ws + pos;
    pos += bytes;
    return p;
  };

  _Float16* Xh = (_Float16*)alloc((size_t)BB * SS * DIN * 2);

  int wHn[24], wKn[24], wNn[24];
  _Float16* Wt[24];
  for (int i = 0; i < 24; ++i) {
    if (i < 6)       { wHn[i] = HH; wKn[i] = DIN; wNn[i] = 64; }
    else if (i < 13) { wHn[i] = HH; wKn[i] = 64;  wNn[i] = 32; }
    else if (i < 22) { wHn[i] = HH; wKn[i] = 32;  wNn[i] = 64; }
    else             { wHn[i] = 1;  wKn[i] = DIN; wNn[i] = DIN; }
    Wt[i] = (_Float16*)alloc((size_t)wHn[i] * wKn[i] * wNn[i] * 2);
  }

  _Float16* stH = (_Float16*)alloc((size_t)BB * SLEN * DIN * 2);
  float*    stF = (float*)   alloc((size_t)BB * SLEN * DIN * 4);

  const long h64  = (long)LL * 64, bh64 = (long)HH * h64;
  const long h32  = (long)LL * 32, bh32 = (long)HH * h32;

  _Float16* Q1  = (_Float16*)alloc((size_t)BB * bh64 * 2);
  _Float16* K1  = (_Float16*)alloc((size_t)BB * bh64 * 2);
  _Float16* V1t = (_Float16*)alloc((size_t)BB * bh64 * 2);
  _Float16* A1  = (_Float16*)alloc((size_t)BB * bh64 * 2);
  _Float16* Q2  = (_Float16*)alloc((size_t)BB * bh32 * 2);
  _Float16* K2  = (_Float16*)alloc((size_t)BB * bh32 * 2);
  _Float16* V2t = (_Float16*)alloc((size_t)BB * bh32 * 2);
  _Float16* A2  = (_Float16*)alloc((size_t)BB * bh32 * 2);
  _Float16* Q3  = (_Float16*)alloc((size_t)BB * bh64 * 2);
  _Float16* K3  = (_Float16*)alloc((size_t)BB * bh64 * 2);
  _Float16* V3t = (_Float16*)alloc((size_t)BB * bh64 * 2);
  _Float16* A3  = (_Float16*)alloc((size_t)BB * LL * DIN * 2);

  dim3 blk(256), gblk(256), ablk(128);
  auto cvt = [&](const float* s, _Float16* d, long n, long mod) {
    k_cvt<<<dim3((unsigned)((n + 255) / 256)), blk, 0, stream>>>(s, d, n, mod);
  };
  auto gemm = [&](const _Float16* A, long aB, long aH, int lda,
                  const _Float16* Wp, long wHs, int M, int N, int K,
                  void* C, long cB, long cH, int ldc, int roff, int mode,
                  int gy, int gz) {
    const int mBlocks = M / 128;
    const int gx = (K >= 256) ? mBlocks * (N / 64) : mBlocks;
    dim3 grid((unsigned)gx, (unsigned)gy, (unsigned)gz);
    k_gemm<<<grid, gblk, 0, stream>>>(A, aB, aH, lda, Wp, wHs, N, K, mBlocks,
                                      C, cB, cH, ldc, roff, mode);
  };

  // ---- once-per-call setup ----
  cvt(x, Xh, (long)BB * SS * DIN, (long)BB * SS * DIN);
  for (int i = 0; i < 24; ++i) {
    long tot = (long)wHn[i] * wKn[i] * wNn[i];
    k_wT<<<dim3((unsigned)((tot + 255) / 256)), blk, 0, stream>>>(
        W[i], Wt[i], wHn[i], wKn[i], wNn[i]);
  }
  cvt(st0, stH, (long)BB * SLEN * DIN, (long)SLEN * DIN);  // broadcast to B

  const long stStride = (long)SLEN * DIN;
  struct Seg { int aoff, M, croff; };
  const Seg sgs[3] = { {0, 128, 0}, {128, 512, 128}, {640, 128, 640} };
  const int dq[3] = {iWdq_ss, iWdk, iWdq_se};
  const int dk[3] = {iWdk_ss, iWdk, iWdk_se};
  const int dv[3] = {iWdv_ss, iWdk, iWdv_se};
  const int uq[3] = {iWuq_ss, iWuq, iWuq_se};
  const int uk[3] = {iWuk_ss, iWuk, iWuk_se};
  const int uv[3] = {iWuv_ss, iWuv, iWuv_se};

  // ---- sequential scan over segments ----
  for (int t = 0; t < NSEG; ++t) {
    // state projections into both ends of the concat (rows 0..127 and 640..767)
    for (int e = 0; e < 2; ++e) {
      const int ro = e ? 640 : 0;
      gemm(stH, stStride, 0, DIN, Wt[iWq_st], (long)64 * DIN, 128, 64, DIN,
           Q1, bh64, h64, 64, ro, 0, HH, BB);
      gemm(stH, stStride, 0, DIN, Wt[iWk_st], (long)64 * DIN, 128, 64, DIN,
           K1, bh64, h64, 64, ro, 0, HH, BB);
      gemm(stH, stStride, 0, DIN, Wt[iWv_st], (long)64 * DIN, 128, 64, DIN,
           V1t, bh64, h64, LL, ro, 1, HH, BB);
    }
    // x projections into rows 128..639
    const _Float16* Xseg = Xh + (long)t * SEGL * DIN;
    gemm(Xseg, (long)SS * DIN, 0, DIN, Wt[iWq], (long)64 * DIN, 512, 64, DIN,
         Q1, bh64, h64, 64, 128, 0, HH, BB);
    gemm(Xseg, (long)SS * DIN, 0, DIN, Wt[iWk], (long)64 * DIN, 512, 64, DIN,
         K1, bh64, h64, 64, 128, 0, HH, BB);
    gemm(Xseg, (long)SS * DIN, 0, DIN, Wt[iWv], (long)64 * DIN, 512, 64, DIN,
         V1t, bh64, h64, LL, 128, 1, HH, BB);
    // attention 1 (d=64)
    k_attn<64><<<dim3(LL / 64, HH, BB), ablk, 0, stream>>>(
        Q1, K1, V1t, h64, A1, bh64, h64, 64);
    // down projections (64 -> 32); middle uses shared i_m = a_m @ Wdk
    for (int s = 0; s < 3; ++s) {
      const _Float16* Ao = A1 + (long)sgs[s].aoff * 64;
      gemm(Ao, bh64, h64, 64, Wt[dq[s]], (long)32 * 64, sgs[s].M, 32, 64,
           Q2, bh32, h32, 32, sgs[s].croff, 0, HH, BB);
      gemm(Ao, bh64, h64, 64, Wt[dk[s]], (long)32 * 64, sgs[s].M, 32, 64,
           K2, bh32, h32, 32, sgs[s].croff, 0, HH, BB);
      gemm(Ao, bh64, h64, 64, Wt[dv[s]], (long)32 * 64, sgs[s].M, 32, 64,
           V2t, bh32, h32, LL, sgs[s].croff, 1, HH, BB);
    }
    // attention 2 (d=32)
    k_attn<32><<<dim3(LL / 64, HH, BB), ablk, 0, stream>>>(
        Q2, K2, V2t, h32, A2, bh32, h32, 32);
    // up projections (32 -> 64)
    for (int s = 0; s < 3; ++s) {
      const _Float16* Ao = A2 + (long)sgs[s].aoff * 32;
      gemm(Ao, bh32, h32, 32, Wt[uq[s]], (long)64 * 32, sgs[s].M, 64, 32,
           Q3, bh64, h64, 64, sgs[s].croff, 0, HH, BB);
      gemm(Ao, bh32, h32, 32, Wt[uk[s]], (long)64 * 32, sgs[s].M, 64, 32,
           K3, bh64, h64, 64, sgs[s].croff, 0, HH, BB);
      gemm(Ao, bh32, h32, 32, Wt[uv[s]], (long)64 * 32, sgs[s].M, 64, 32,
           V3t, bh64, h64, LL, sgs[s].croff, 1, HH, BB);
    }
    // attention 3 (d=64) -> A3 laid out (B, 768, H*64) for the output GEMMs
    k_attn<64><<<dim3(LL / 64, HH, BB), ablk, 0, stream>>>(
        Q3, K3, V3t, h64, A3, (long)LL * DIN, 64, DIN);
    // y = a3_m @ Wout  (f32, straight into d_out)
    gemm(A3 + (long)128 * DIN, (long)LL * DIN, 0, DIN, Wt[iWout], 0,
         512, DIN, DIN, out + (long)t * SEGL * DIN, (long)SS * DIN, 0, DIN,
         0, 2, 1, BB);
    // new_st = a3_e @ Wout_st (last segment writes the d_out tail directly)
    float* stDst = (t < NSEG - 1) ? stF : (out + (long)BB * SS * DIN);
    gemm(A3 + (long)640 * DIN, (long)LL * DIN, 0, DIN, Wt[iWout_st], 0,
         128, DIN, DIN, stDst, stStride, 0, DIN, 0, 2, 1, BB);
    if (t < NSEG - 1)
      cvt(stF, stH, (long)BB * SLEN * DIN, (long)BB * SLEN * DIN);
  }
}
